// LieRotationAttention_33698313404507
// MI455X (gfx1250) — compile-verified, matrix operand order checked by reading, not run on hardware
//
#include <hip/hip_runtime.h>
#include <hip/hip_bf16.h>
#include <math.h>

#define HIDDEN 768
#define NHEADS 12
#define HDIM   64
#define BATCH  4
#define SEQ    1024
#define BS     (BATCH*SEQ)

typedef _Float16 v8h  __attribute__((ext_vector_type(8)));
typedef _Float16 v16h __attribute__((ext_vector_type(16)));
typedef float    v8f  __attribute__((ext_vector_type(8)));

union H16 { v16h v; v8h h[2]; };

// Build a 16-half WMMA operand from two contiguous 8-half (16B) runs.
__device__ __forceinline__ v16h load_tile16(const _Float16* p0, const _Float16* p1) {
    H16 u;
    u.h[0] = *(const v8h*)p0;
    u.h[1] = *(const v8h*)p1;
    return u.v;
}

// CDNA5 async copy: global -> LDS, 16 bytes per lane, tracked by ASYNCcnt.
__device__ __forceinline__ void async_copy_b128(const _Float16* g, _Float16* lds) {
    unsigned ldsoff = (unsigned)(uintptr_t)lds;            // LDS segment offset (low 32 bits)
    unsigned long long ga = (unsigned long long)(uintptr_t)g;
    asm volatile("global_load_async_to_lds_b128 %0, %1, off"
                 :: "v"(ldsoff), "v"(ga) : "memory");
}
__device__ __forceinline__ void wait_async0() {
    asm volatile("s_wait_asynccnt 0" ::: "memory");
}

#define WMMA_F16(a, b, c) \
    __builtin_amdgcn_wmma_f32_16x16x32_f16(false, (a), false, (b), (short)0, (c), false, false)

// ---------------- Kernel 1: SE(3) exp map ----------------
__global__ void se3_kernel(const float* __restrict__ rp,
                           float* __restrict__ Rm, float* __restrict__ Tm) {
    int s = blockIdx.x * blockDim.x + threadIdx.x;
    if (s >= SEQ) return;
    float wx = rp[s*6+0], wy = rp[s*6+1], wz = rp[s*6+2];
    float vx = rp[s*6+3], vy = rp[s*6+4], vz = rp[s*6+5];
    float th = sqrtf(wx*wx + wy*wy + wz*wz);
    th = fmaxf(th, 1e-8f);
    float inv = 1.0f / th;
    float ux = wx*inv, uy = wy*inv, uz = wz*inv;
    float st = sinf(th), ct = cosf(th);
    float A = st, B = 1.0f - ct;
    float xx = ux*ux, yy = uy*uy, zz = uz*uz;
    float xy = ux*uy, xz = ux*uz, yz = uy*uz;
    Rm[s*9+0] = 1.0f + B*(xx-1.0f); Rm[s*9+1] = -A*uz + B*xy;      Rm[s*9+2] =  A*uy + B*xz;
    Rm[s*9+3] =  A*uz + B*xy;      Rm[s*9+4] = 1.0f + B*(yy-1.0f); Rm[s*9+5] = -A*ux + B*yz;
    Rm[s*9+6] = -A*uy + B*xz;      Rm[s*9+7] =  A*ux + B*yz;       Rm[s*9+8] = 1.0f + B*(zz-1.0f);
    float c1 = (1.0f - ct) / (th*th);
    float c2 = (th - st) / (th*th*th);
    float V00 = 1.0f + c2*(xx-1.0f), V01 = -c1*uz + c2*xy,      V02 =  c1*uy + c2*xz;
    float V10 =  c1*uz + c2*xy,      V11 = 1.0f + c2*(yy-1.0f), V12 = -c1*ux + c2*yz;
    float V20 = -c1*uy + c2*xz,      V21 =  c1*ux + c2*yz,      V22 = 1.0f + c2*(zz-1.0f);
    Tm[s*3+0] = V00*vx + V01*vy + V02*vz;
    Tm[s*3+1] = V10*vx + V11*vy + V12*vz;
    Tm[s*3+2] = V20*vx + V21*vy + V22*vz;
}

// ---------------- Kernel 2: rotate x, emit f16 ----------------
__global__ void rotate_kernel(const float* __restrict__ x, const float* __restrict__ Rm,
                              const float* __restrict__ Tm, _Float16* __restrict__ xrh) {
    int idx = blockIdx.x * blockDim.x + threadIdx.x;   // bs*256 + n
    if (idx >= BS*256) return;
    int n  = idx & 255;
    int bs = idx >> 8;
    int s  = bs & (SEQ-1);
    const float* xp = x + (size_t)bs*HIDDEN + n*3;
    float x0 = xp[0], x1 = xp[1], x2 = xp[2];
    const float* Rp = Rm + s*9;
    const float* Tp = Tm + s*3;
    float r0 = Rp[0]*x0 + Rp[1]*x1 + Rp[2]*x2 + Tp[0];
    float r1 = Rp[3]*x0 + Rp[4]*x1 + Rp[5]*x2 + Tp[1];
    float r2 = Rp[6]*x0 + Rp[7]*x1 + Rp[8]*x2 + Tp[2];
    _Float16* op = xrh + (size_t)bs*HIDDEN + n*3;
    op[0] = (_Float16)r0; op[1] = (_Float16)r1; op[2] = (_Float16)r2;
}

// ---------------- Kernel 3: f32 -> f16 convert ----------------
__global__ void cvt_kernel(const float* __restrict__ src, _Float16* __restrict__ dst, int n) {
    int i = blockIdx.x * blockDim.x + threadIdx.x;
    if (i < n) dst[i] = (_Float16)src[i];
}

// ---------------- Kernel 4: LDS-tiled, double-buffered WMMA GEMM ----------------
// C[m][o] = sum_h A[m][h] * W[o][h];  M=4096, N=768, K=768.
// 8 waves per block -> 64x64 tile; K in 64-chunks, ping-pong LDS staging filled
// by async-to-LDS b128 copies; copy(i+1) overlaps compute(i).
// mode 0: f16 out [b,h,s,d] | mode 1: f16 out [b,h,d,s] | mode 2: f32 [m][o]
__global__ void __launch_bounds__(256)
gemm_wmma(const _Float16* __restrict__ A, const _Float16* __restrict__ W,
          void* __restrict__ dst, int mode) {
    __shared__ __align__(16) _Float16 As[2][64][64];   // 16 KB
    __shared__ __align__(16) _Float16 Bs[2][64][64];   // 16 KB
    int n0 = blockIdx.x * 64;
    int m0 = blockIdx.y * 64;
    int t  = threadIdx.x;
    int w  = t >> 5;
    int L  = t & 31;
    int lg = L >> 4;           // lane group: selects K-half per ISA operand layout
    int ln = L & 15;           // M (for A) / N (for B) index within 16
    int wm = (w & 3) * 16;     // wave's row strip in tile
    int wn = (w >> 2) * 32;    // wave's col strip in tile

    // staging map: each thread copies 2x16B for A and 2x16B for B per chunk
    int e0 = t, e1 = t + 256;
    int r0 = e0 >> 3, c0 = (e0 & 7) * 8;
    int r1 = e1 >> 3, c1 = (e1 & 7) * 8;

    const int NCH = HIDDEN / 64;          // 12 chunks
    // prologue: chunk 0 into buffer 0
    async_copy_b128(A + (size_t)(m0 + r0)*HIDDEN + c0, &As[0][r0][c0]);
    async_copy_b128(A + (size_t)(m0 + r1)*HIDDEN + c1, &As[0][r1][c1]);
    async_copy_b128(W + (size_t)(n0 + r0)*HIDDEN + c0, &Bs[0][r0][c0]);
    async_copy_b128(W + (size_t)(n0 + r1)*HIDDEN + c1, &Bs[0][r1][c1]);

    v8f acc0 = {}, acc1 = {};
    for (int i = 0; i < NCH; ++i) {
        int cur = i & 1;
        wait_async0();                     // my share of chunk i landed
        __syncthreads();                   // everyone's landed; compute(i-1) done everywhere
        if (i + 1 < NCH) {                 // overlap copy(i+1) with compute(i)
            int kc = (i + 1) * 64, nb = cur ^ 1;
            async_copy_b128(A + (size_t)(m0 + r0)*HIDDEN + kc + c0, &As[nb][r0][c0]);
            async_copy_b128(A + (size_t)(m0 + r1)*HIDDEN + kc + c1, &As[nb][r1][c1]);
            async_copy_b128(W + (size_t)(n0 + r0)*HIDDEN + kc + c0, &Bs[nb][r0][c0]);
            async_copy_b128(W + (size_t)(n0 + r1)*HIDDEN + kc + c1, &Bs[nb][r1][c1]);
        }
        #pragma unroll
        for (int kk = 0; kk < 64; kk += 32) {
            v16h a  = load_tile16(&As[cur][wm + ln][kk + lg*8],      &As[cur][wm + ln][kk + 16 + lg*8]);
            v16h b0 = load_tile16(&Bs[cur][wn + ln][kk + lg*8],      &Bs[cur][wn + ln][kk + 16 + lg*8]);
            v16h b1 = load_tile16(&Bs[cur][wn + 16 + ln][kk + lg*8], &Bs[cur][wn + 16 + ln][kk + 16 + lg*8]);
            acc0 = WMMA_F16(a, b0, acc0);
            acc1 = WMMA_F16(a, b1, acc1);
        }
    }

    int mbase = m0 + wm + lg*8;
    #pragma unroll
    for (int nt = 0; nt < 2; ++nt) {
        v8f acc = nt ? acc1 : acc0;
        int o = n0 + wn + nt*16 + ln;
        if (mode == 2) {
            float* out = (float*)dst;
            #pragma unroll
            for (int r = 0; r < 8; ++r)
                out[(size_t)(mbase + r) * HIDDEN + o] = acc[r];
        } else {
            _Float16* out = (_Float16*)dst;
            int head = o >> 6, d = o & 63;
            #pragma unroll
            for (int r = 0; r < 8; ++r) {
                int m = mbase + r;
                int b_ = m >> 10, s = m & (SEQ-1);
                int bh = b_ * NHEADS + head;
                if (mode == 0)
                    out[(((size_t)bh << 10) + s) * HDIM + d] = (_Float16)acc[r];
                else
                    out[(((size_t)bh * HDIM + d) << 10) + s] = (_Float16)acc[r];
            }
        }
    }
}

// ---------------- Kernel 5: fused causal attention ----------------
// 4 waves (128 threads) handle 64 query rows of one (b,head); K/V tiles are
// shared across waves via double-buffered async-to-LDS staging.
// Dynamic LDS: 4 x (16x1024) f16 score rows (128 KB) + 2 x 4 KB staging.
#define ATTN_SC_HALFS  (4*16*SEQ)
#define ATTN_ST_HALFS  (2*2048)
#define ATTN_LDS_BYTES ((ATTN_SC_HALFS + ATTN_ST_HALFS)*2)

__device__ __forceinline__ void stage_k(const _Float16* __restrict__ Kt, size_t bhK,
                                        int k0, _Float16* stbuf, int t) {
    #pragma unroll
    for (int i = 0; i < 2; ++i) {              // 32 rows x 64 halfs
        int e = t + i*128;
        int r = e >> 3, c8 = (e & 7) * 8;
        async_copy_b128(Kt + bhK + (size_t)(k0 + r)*HDIM + c8, stbuf + r*64 + c8);
    }
}
__device__ __forceinline__ void stage_v(const _Float16* __restrict__ Vt, size_t bhV,
                                        int s0, _Float16* stbuf, int t) {
    #pragma unroll
    for (int i = 0; i < 2; ++i) {              // 64 d-rows x 32 halfs
        int e = t + i*128;
        int d = e >> 2, c4 = (e & 3) * 8;
        async_copy_b128(Vt + bhV + ((size_t)d << 10) + s0 + c4, stbuf + d*32 + c4);
    }
}

__global__ void __launch_bounds__(128)
attn_kernel(const _Float16* __restrict__ Q, const _Float16* __restrict__ Kt,
            const _Float16* __restrict__ Vt, float* __restrict__ attn_g,
            _Float16* __restrict__ xout) {
    extern __shared__ __align__(16) _Float16 smem[];
    int qb = blockIdx.x;
    int bh = blockIdx.y;
    int b  = bh / NHEADS, head = bh % NHEADS;
    int t  = threadIdx.x;
    int w  = t >> 5;
    int L  = t & 31;
    int lg = L >> 4, ln = L & 15;
    int q0  = qb * 64;
    int q0w = q0 + w * 16;

    _Float16* scw = smem + (size_t)w * 16 * SEQ;   // this wave's score rows
    _Float16* st  = smem + (size_t)ATTN_SC_HALFS;  // shared 2-buffer staging

    size_t bhK = ((size_t)bh << 10) * HDIM;
    size_t bhV = ((size_t)bh * HDIM) << 10;

    // hoist Q fragment (16 rows x 64) into registers for the whole kernel
    const _Float16* qrow = Q + bhK + (size_t)(q0w + ln) * HDIM;
    v16h aq0 = load_tile16(qrow + lg*8,      qrow + 16 + lg*8);
    v16h aq1 = load_tile16(qrow + 32 + lg*8, qrow + 48 + lg*8);

    int ntiles  = (q0 + 64) >> 5;          // 32-row K/V tiles for the block
    int kcols_w = q0w + 16;                // this wave's causal width
    int tilesW  = (kcols_w + 31) >> 5;     // tiles this wave computes
    int kcs     = tilesW << 5;             // stored score width (-inf padded)

    // ---- pass 1: scores = Q K^T / 8 (K tiles shared via LDS) ----
    stage_k(Kt, bhK, 0, st, t);
    for (int i = 0; i < ntiles; ++i) {
        int cur = i & 1;
        wait_async0();
        __syncthreads();
        if (i + 1 < ntiles) stage_k(Kt, bhK, (i+1)*32, st + (cur^1)*2048, t);
        if (i < tilesW) {
            const _Float16* Ks = st + cur*2048;
            int k0 = i << 5;
            #pragma unroll
            for (int ns = 0; ns < 2; ++ns) {
                const _Float16* krow = Ks + (ns*16 + ln)*64;
                v8f acc = {};
                v16h b0 = load_tile16(krow + lg*8,      krow + 16 + lg*8);
                acc = WMMA_F16(aq0, b0, acc);
                v16h b1 = load_tile16(krow + 32 + lg*8, krow + 48 + lg*8);
                acc = WMMA_F16(aq1, b1, acc);
                #pragma unroll
                for (int r = 0; r < 8; ++r) {
                    int row = lg*8 + r;
                    int qg  = q0w + row;
                    int kk  = k0 + ns*16 + ln;
                    float v = acc[r] * 0.125f;     // 1/sqrt(64)
                    scw[row*SEQ + kk] = (kk <= qg) ? (_Float16)v : (_Float16)(-INFINITY);
                }
            }
        }
    }

    // ---- softmax over [0,kcs): 2 lanes per row (-inf slots -> p=0 pad) ----
    int row = ln, half = lg;
    int C2 = kcs >> 1;
    int cbeg = half * C2, cend = cbeg + C2;
    float mx = -INFINITY;
    for (int c = cbeg; c < cend; ++c) mx = fmaxf(mx, (float)scw[row*SEQ + c]);
    mx = fmaxf(mx, __shfl_xor(mx, 16, 32));
    float sum = 0.0f;
    for (int c = cbeg; c < cend; ++c) sum += __expf((float)scw[row*SEQ + c] - mx);
    sum += __shfl_xor(sum, 16, 32);
    float invs = 1.0f / sum;
    int qg = q0w + row;
    float* arow = attn_g + (((size_t)bh << 10) + qg) * SEQ;
    for (int c = cbeg; c < cend; ++c) {
        float p = __expf((float)scw[row*SEQ + c] - mx) * invs;
        scw[row*SEQ + c] = (_Float16)p;
        arow[c] = p;
    }
    // zero the causal upper triangle in the attn output (d_out is poisoned)
    int rest = SEQ - kcs;
    int Z2 = rest >> 1;
    for (int c = kcs + half*Z2; c < kcs + (half+1)*Z2; ++c) arow[c] = 0.0f;

    // ---- pass 2: out = P V (V tiles shared via LDS, from Vt [bh][d][s]) ----
    __syncthreads();                       // all waves done reading K staging
    stage_v(Vt, bhV, 0, st, t);
    v8f oacc[4] = {};
    for (int i = 0; i < ntiles; ++i) {
        int cur = i & 1;
        wait_async0();
        __syncthreads();
        if (i + 1 < ntiles) stage_v(Vt, bhV, (i+1)*32, st + (cur^1)*2048, t);
        if (i < tilesW) {
            const _Float16* Vs = st + cur*2048;
            const _Float16* prow = scw + ln*SEQ + (i << 5);
            v16h a = load_tile16(prow + lg*8, prow + 16 + lg*8);
            #pragma unroll
            for (int dt = 0; dt < 4; ++dt) {
                const _Float16* vr = Vs + (dt*16 + ln)*32;
                v16h bm = load_tile16(vr + lg*8, vr + 16 + lg*8);
                oacc[dt] = WMMA_F16(a, bm, oacc[dt]);
            }
        }
    }
    #pragma unroll
    for (int dt = 0; dt < 4; ++dt) {
        #pragma unroll
        for (int r = 0; r < 8; ++r) {
            int rr = lg*8 + r;
            int q  = q0w + rr;
            int d  = dt*16 + ln;
            xout[((size_t)b * SEQ + q) * HIDDEN + head*HDIM + d] = (_Float16)oacc[dt][r];
        }
    }
}

// ---------------- host launcher ----------------
extern "C" void kernel_launch(void* const* d_in, const int* in_sizes, int n_in,
                              void* d_out, int out_size, void* d_ws, size_t ws_size,
                              hipStream_t stream) {
    const float* x  = (const float*)d_in[0];
    const float* Wq = (const float*)d_in[1];
    const float* Wk = (const float*)d_in[2];
    const float* Wv = (const float*)d_in[3];
    const float* Wo = (const float*)d_in[4];
    const float* rp = (const float*)d_in[5];

    char* ws = (char*)d_ws;
    size_t off = 0;
    float*    Rm   = (float*)(ws + off);  off += (size_t)SEQ*9*4;
    float*    Tm   = (float*)(ws + off);  off += (size_t)SEQ*3*4;
    off = (off + 255) & ~(size_t)255;
    _Float16* xrh  = (_Float16*)(ws + off); off += (size_t)BS*HIDDEN*2;
    _Float16* wqh  = (_Float16*)(ws + off); off += (size_t)HIDDEN*HIDDEN*2;
    _Float16* wkh  = (_Float16*)(ws + off); off += (size_t)HIDDEN*HIDDEN*2;
    _Float16* wvh  = (_Float16*)(ws + off); off += (size_t)HIDDEN*HIDDEN*2;
    _Float16* woh  = (_Float16*)(ws + off); off += (size_t)HIDDEN*HIDDEN*2;
    _Float16* Qh   = (_Float16*)(ws + off); off += (size_t)BATCH*NHEADS*SEQ*HDIM*2;
    _Float16* Kh   = (_Float16*)(ws + off); off += (size_t)BATCH*NHEADS*SEQ*HDIM*2;
    _Float16* Vth  = (_Float16*)(ws + off); off += (size_t)BATCH*NHEADS*SEQ*HDIM*2;
    _Float16* xouth= (_Float16*)(ws + off); off += (size_t)BS*HIDDEN*2;

    float* out_p  = (float*)d_out;
    float* attn_p = out_p + (size_t)BS*HIDDEN;

    // allow >64KB dynamic LDS for the attention kernel (WGP supports 320KB)
    (void)hipFuncSetAttribute((const void*)attn_kernel,
                              hipFuncAttributeMaxDynamicSharedMemorySize,
                              ATTN_LDS_BYTES);

    se3_kernel<<<dim3(SEQ/256), dim3(256), 0, stream>>>(rp, Rm, Tm);
    rotate_kernel<<<dim3(BS*256/256), dim3(256), 0, stream>>>(x, Rm, Tm, xrh);

    const int WN = HIDDEN*HIDDEN;
    cvt_kernel<<<dim3((WN+255)/256), dim3(256), 0, stream>>>(Wq, wqh, WN);
    cvt_kernel<<<dim3((WN+255)/256), dim3(256), 0, stream>>>(Wk, wkh, WN);
    cvt_kernel<<<dim3((WN+255)/256), dim3(256), 0, stream>>>(Wv, wvh, WN);
    cvt_kernel<<<dim3((WN+255)/256), dim3(256), 0, stream>>>(Wo, woh, WN);

    dim3 gg(HIDDEN/64, BS/64);   // 12 x 64 tiles of 64x64
    gemm_wmma<<<gg, 256, 0, stream>>>(xrh, wqh, (void*)Qh, 0);
    gemm_wmma<<<gg, 256, 0, stream>>>(xrh, wkh, (void*)Kh, 0);
    gemm_wmma<<<gg, 256, 0, stream>>>(xrh, wvh, (void*)Vth, 1);

    dim3 ga(SEQ/64, BATCH*NHEADS);  // 16 x 48 blocks, 128 threads each
    attn_kernel<<<ga, 128, ATTN_LDS_BYTES, stream>>>(Qh, Kh, Vth, attn_p, xouth);

    gemm_wmma<<<gg, 256, 0, stream>>>(xouth, woh, (void*)d_out, 2);
}
